// SWG_14628658610514
// MI455X (gfx1250) — matrix operand order repaired and checked by reading, not compiled, not run on hardware
//
#include <hip/hip_runtime.h>
#include <math.h>

// ---------------- problem constants (match reference setup_inputs) ----------
#define NRN 30000     // r nodes
#define NCN 100000    // c nodes
#define FIN 64
#define HID 128
#define NEDGE 250000
#define NLBL 100000
#define NHEAD 4
#define DHEAD 32
#define NLAYER 2

typedef __attribute__((ext_vector_type(16))) _Float16 v16h;
typedef __attribute__((ext_vector_type(8)))  float    v8f;

// ---------------------------------------------------------------------------
// helpers
// ---------------------------------------------------------------------------
__device__ inline float gelu_exact(float x) {
  return 0.5f * x * (1.0f + erff(x * 0.70710678118654752f));
}

__device__ inline void atomicMaxF(float* addr, float val) {
  if (val >= 0.0f) atomicMax((int*)addr, __float_as_int(val));
  else             atomicMin((unsigned int*)addr, __float_as_uint(val));
}

// ---------------------------------------------------------------------------
// Generic WMMA GEMM:  C[M,N] = epi( (preGelu? gelu(A) : A) @ W + bias )
//   epi 0: none   epi 1: relu   epi 2: sg*o + (1-sg)*X,  sg = sigmoid(skip[0])
//
// Fragments are built directly from global memory using the CDNA5 VGPR
// layouts (ISA 7.12.2):
//   A (16x32, f16): lane L -> row m = L&15, hi = L>>4; elements are two
//     contiguous 8-float runs at k0+8*hi and k0+16+8*hi  -> 4x b128 loads.
//   B (32x16, f16): lane L -> col n = L&15; element j -> K = k0+8*hi+j (j<8),
//     k0+16+8*hi+(j-8) (j>=8)
//     (lanes 0-15 / 16-31 each read one contiguous 64B row segment per j;
//      all W matrices here are small and L2-resident).
//   C/D (16x16, f32): lane L -> col n = L&15, VGPR r -> row r + 8*(L>>4).
//
// Each wave owns a 16 x (16*NT) output strip: one A fragment is reused
// across NT B fragments -> NT v_wmma per K step (fully unrolled).
// Block = 128 threads = 4 waves tiling M; wave-level M guard (uniform; no
// barriers are used, EXEC is all-ones inside every WMMA).
// Requirements (hold at every call site): M % 16 == 0, K % 32 == 0,
// N % (16*NT) == 0, lda % 4 == 0, 16B-aligned base pointers.
// ---------------------------------------------------------------------------
template <int NT>
__global__ __launch_bounds__(128)
void wmma_gemm_kernel(const float* __restrict__ A, int lda,
                      const float* __restrict__ W, int ldb,
                      const float* __restrict__ bias,
                      float* __restrict__ C, int ldc,
                      const float* __restrict__ X, int ldx,
                      const float* __restrict__ skip,
                      int M, int N, int K, int preGelu, int epi) {
  const int wave = threadIdx.x >> 5;
  const int lane = threadIdx.x & 31;
  const int m0 = blockIdx.x * 64 + wave * 16;
  const int n0 = blockIdx.y * (16 * NT);
  if (m0 >= M) return;  // uniform per wave

  const int hi   = lane >> 4;
  const int mrow = m0 + (lane & 15);
  const int ncol = n0 + (lane & 15);

  v8f acc[NT];
#pragma unroll
  for (int t = 0; t < NT; ++t) acc[t] = (v8f){};

  for (int k0 = 0; k0 < K; k0 += 32) {
    if (k0 + 32 < K)
      __builtin_prefetch(A + (size_t)mrow * lda + k0 + 32, 0, 1);

    // ---- A fragment: two contiguous 8-float runs, vectorized b128 loads ----
    const float* ap = A + (size_t)mrow * lda + k0 + 8 * hi;
    float4 r0 = *(const float4*)(ap);
    float4 r1 = *(const float4*)(ap + 4);
    float4 r2 = *(const float4*)(ap + 16);
    float4 r3 = *(const float4*)(ap + 20);
    float av[16] = {r0.x, r0.y, r0.z, r0.w, r1.x, r1.y, r1.z, r1.w,
                    r2.x, r2.y, r2.z, r2.w, r3.x, r3.y, r3.z, r3.w};
    v16h a;
#pragma unroll
    for (int j = 0; j < 16; ++j) {
      float v = av[j];
      if (preGelu) v = gelu_exact(v);
      a[j] = (_Float16)v;
    }

    // ---- NT x { B fragment + WMMA } ----
#pragma unroll
    for (int t = 0; t < NT; ++t) {
      const float* wp = W + (size_t)k0 * ldb + (ncol + t * 16);
      v16h b;
#pragma unroll
      for (int j = 0; j < 16; ++j) b[j] = (_Float16)wp[(size_t)((j < 8) ? (8 * hi + j) : (16 + 8 * hi + (j - 8))) * ldb];
      acc[t] = __builtin_amdgcn_wmma_f32_16x16x32_f16(
          false, a, false, b, (short)0, acc[t], false, false);
    }
  }

  // ---- epilogue ----
  float sg = 0.0f;
  if (epi == 2) sg = 1.0f / (1.0f + expf(-skip[0]));
#pragma unroll
  for (int t = 0; t < NT; ++t) {
    const int col = ncol + t * 16;
    const float bv = bias ? bias[col] : 0.0f;
#pragma unroll
    for (int r = 0; r < 8; ++r) {
      const int row = m0 + r + 8 * hi;
      float o = acc[t][r] + bv;
      if (epi == 1)      o = fmaxf(o, 0.0f);
      else if (epi == 2) o = sg * o + (1.0f - sg) * X[(size_t)row * ldx + col];
      C[(size_t)row * ldc + col] = o;
    }
  }
}

// ---------------------------------------------------------------------------
// elementwise / edge kernels (L2-resident gather/scatter segment softmax)
// ---------------------------------------------------------------------------
__global__ void fill_kernel(float* __restrict__ p, float v, long n) {
  long i = (long)blockIdx.x * blockDim.x + threadIdx.x;
  long stride = (long)gridDim.x * blockDim.x;
  for (; i < n; i += stride) p[i] = v;
}

// alpha[e,h] = <q_dst[dst,h,:], krel[src,h,:]> * p[h] / sqrt(D);  seg-max -> m
__global__ void edge_logits_kernel(const int* __restrict__ src,
                                   const int* __restrict__ dst,
                                   const float* __restrict__ krel,     // [Ns,128]
                                   const float* __restrict__ kqv_dst,  // [Nd,384]
                                   const float* __restrict__ prel,     // [4]
                                   float* __restrict__ alpha,          // [E,4]
                                   float* __restrict__ mbuf,           // [Nd,4]
                                   int E) {
  int idx = blockIdx.x * blockDim.x + threadIdx.x;
  int e = idx >> 2, h = idx & 3;
  if (e >= E) return;
  int s = src[e], d = dst[e];
  const float* q = kqv_dst + (size_t)d * 384 + 128 + h * DHEAD;
  const float* k = krel + (size_t)s * HID + h * DHEAD;
  float acc = 0.0f;
#pragma unroll
  for (int j = 0; j < DHEAD; ++j) acc += q[j] * k[j];
  acc *= prel[h] * 0.17677669529663687f;  // 1/sqrt(32)
  alpha[(size_t)e * NHEAD + h] = acc;
  atomicMaxF(mbuf + (size_t)d * NHEAD + h, acc);
}

// alpha <- exp(alpha - m[dst]) ; seg-sum -> s
__global__ void edge_exp_kernel(const int* __restrict__ dst,
                                float* __restrict__ alpha,
                                const float* __restrict__ mbuf,
                                float* __restrict__ sbuf, int E) {
  int idx = blockIdx.x * blockDim.x + threadIdx.x;
  int e = idx >> 2, h = idx & 3;
  if (e >= E) return;
  int d = dst[e];
  float ex = expf(alpha[(size_t)e * NHEAD + h] - mbuf[(size_t)d * NHEAD + h]);
  alpha[(size_t)e * NHEAD + h] = ex;
  atomicAdd(sbuf + (size_t)d * NHEAD + h, ex);
}

// agg[dst,h,:] += vrel[src,h,:] * alpha/(s+eps)
__global__ void edge_scatter_kernel(const int* __restrict__ src,
                                    const int* __restrict__ dst,
                                    const float* __restrict__ alpha,
                                    const float* __restrict__ sbuf,
                                    const float* __restrict__ vrel,  // [Ns,128]
                                    float* __restrict__ agg,         // [Nd,128]
                                    int E) {
  int idx = blockIdx.x * blockDim.x + threadIdx.x;
  int e = idx >> 2, h = idx & 3;
  if (e >= E) return;
  int s = src[e], d = dst[e];
  float w = alpha[(size_t)e * NHEAD + h] /
            (sbuf[(size_t)d * NHEAD + h] + 1e-16f);
  const float* v = vrel + (size_t)s * HID + h * DHEAD;
  float* o = agg + (size_t)d * HID + h * DHEAD;
#pragma unroll
  for (int j = 0; j < DHEAD; ++j) atomicAdd(o + j, v[j] * w);
}

// z[e] = concat(h_r[row[e]], h_c[col[e]])  -> [EL,256]
__global__ void zgather_kernel(const int* __restrict__ row,
                               const int* __restrict__ col,
                               const float* __restrict__ hr,
                               const float* __restrict__ hc,
                               float* __restrict__ z, int E) {
  long i = (long)blockIdx.x * blockDim.x + threadIdx.x;
  long n = (long)E * 256;
  if (i >= n) return;
  int e = (int)(i >> 8), c = (int)(i & 255);
  float v = (c < HID) ? hr[(size_t)row[e] * HID + c]
                      : hc[(size_t)col[e] * HID + (c - HID)];
  z[i] = v;
}

// score[e] = <hmid[e,:], w2> + b2
__global__ void score_kernel(const float* __restrict__ hmid,
                             const float* __restrict__ w2,
                             const float* __restrict__ b2,
                             float* __restrict__ out, int E) {
  int e = blockIdx.x * blockDim.x + threadIdx.x;
  if (e >= E) return;
  float acc = b2[0];
#pragma unroll 8
  for (int i = 0; i < HID; ++i) acc += hmid[(size_t)e * HID + i] * w2[i];
  out[e] = acc;
}

// ---------------------------------------------------------------------------
// host side
// ---------------------------------------------------------------------------
struct ConvP {
  const float *Wkqv_r, *bkqv_r, *Wkqv_c, *bkqv_c;
  const float *a_c2r, *m_c2r, *p_c2r, *a_r2c, *m_r2c, *p_r2c;
  const float *Wout_r, *bout_r, *Wout_c, *bout_c, *skip_r, *skip_c;
};
struct InP {
  const float *x_r, *x_c;
  const float *W_in_r, *b_in_r, *W_in_c, *b_in_c;
  const float *W_mlp1, *b_mlp1, *W_mlp2, *b_mlp2;
  const int *edge_c2r, *edge_r2c, *label_edge;
  ConvP conv[NLAYER];
};

static void map_inputs(void* const* d_in, const int* in_sizes, InP* P) {
  auto F = [&](int i) { return (const float*)d_in[i]; };
  auto I = [&](int i) { return (const int*)d_in[i]; };
  if (in_sizes[0] == NRN * FIN) {
    // insertion-order flattening: x_r, x_c, params{...}, edges
    P->x_r = F(0); P->x_c = F(1);
    P->W_in_r = F(2); P->b_in_r = F(3); P->W_in_c = F(4); P->b_in_c = F(5);
    int k = 6;
    for (int l = 0; l < NLAYER; ++l) {
      ConvP* c = &P->conv[l];
      c->Wkqv_r = F(k + 0);  c->bkqv_r = F(k + 1);
      c->Wkqv_c = F(k + 2);  c->bkqv_c = F(k + 3);
      c->a_c2r  = F(k + 4);  c->m_c2r  = F(k + 5);  c->p_c2r = F(k + 6);
      c->a_r2c  = F(k + 7);  c->m_r2c  = F(k + 8);  c->p_r2c = F(k + 9);
      c->Wout_r = F(k + 10); c->bout_r = F(k + 11);
      c->Wout_c = F(k + 12); c->bout_c = F(k + 13);
      c->skip_r = F(k + 14); c->skip_c = F(k + 15);
      k += 16;
    }
    P->W_mlp1 = F(k + 0); P->b_mlp1 = F(k + 1);
    P->W_mlp2 = F(k + 2); P->b_mlp2 = F(k + 3);
    P->edge_c2r = I(k + 4); P->edge_r2c = I(k + 5); P->label_edge = I(k + 6);
  } else {
    // jax tree_flatten (alphabetical at every dict level)
    P->edge_c2r = I(0); P->edge_r2c = I(1); P->label_edge = I(2);
    P->W_in_c = F(3); P->W_in_r = F(4); P->W_mlp1 = F(5); P->W_mlp2 = F(6);
    P->b_in_c = F(7); P->b_in_r = F(8); P->b_mlp1 = F(9); P->b_mlp2 = F(10);
    int k = 11;
    for (int l = 0; l < NLAYER; ++l) {
      ConvP* c = &P->conv[l];
      c->Wkqv_c = F(k + 0);  c->Wkqv_r = F(k + 1);
      c->Wout_c = F(k + 2);  c->Wout_r = F(k + 3);
      c->a_c2r  = F(k + 4);  c->a_r2c  = F(k + 5);
      c->bkqv_c = F(k + 6);  c->bkqv_r = F(k + 7);
      c->bout_c = F(k + 8);  c->bout_r = F(k + 9);
      c->m_c2r  = F(k + 10); c->m_r2c  = F(k + 11);
      c->p_c2r  = F(k + 12); c->p_r2c  = F(k + 13);
      c->skip_c = F(k + 14); c->skip_r = F(k + 15);
      k += 16;
    }
    P->x_c = F(k); P->x_r = F(k + 1);
  }
}

static inline void gemm(hipStream_t st, const float* A, int lda,
                        const float* W, int ldb, const float* bias,
                        float* C, int ldc, const float* X, int ldx,
                        const float* skip, int M, int N, int K,
                        int preGelu, int epi) {
  const int gx = (M + 63) / 64;
  if (N % 64 == 0) {
    dim3 g(gx, N / 64);
    wmma_gemm_kernel<4><<<g, 128, 0, st>>>(A, lda, W, ldb, bias, C, ldc,
                                           X, ldx, skip, M, N, K, preGelu, epi);
  } else if (N % 32 == 0) {
    dim3 g(gx, N / 32);
    wmma_gemm_kernel<2><<<g, 128, 0, st>>>(A, lda, W, ldb, bias, C, ldc,
                                           X, ldx, skip, M, N, K, preGelu, epi);
  } else {
    dim3 g(gx, N / 16);
    wmma_gemm_kernel<1><<<g, 128, 0, st>>>(A, lda, W, ldb, bias, C, ldc,
                                           X, ldx, skip, M, N, K, preGelu, epi);
  }
}

static inline void fill(hipStream_t st, float* p, float v, long n) {
  int blocks = (int)((n + 255) / 256);
  if (blocks > 65535) blocks = 65535;
  fill_kernel<<<blocks, 256, 0, st>>>(p, v, n);
}

extern "C" void kernel_launch(void* const* d_in, const int* in_sizes, int n_in,
                              void* d_out, int out_size, void* d_ws,
                              size_t ws_size, hipStream_t stream) {
  (void)n_in; (void)out_size; (void)ws_size;
  InP P;
  map_inputs(d_in, in_sizes, &P);

  // ---- workspace arena (f32) ----
  float* ws = (float*)d_ws;
  size_t o = 0;
  float* h_r    = ws + o; o += (size_t)NRN * HID;
  float* h_c    = ws + o; o += (size_t)NCN * HID;
  float* kqv_r  = ws + o; o += (size_t)NRN * 384;
  float* kqv_c  = ws + o; o += (size_t)NCN * 384;
  float* krel_c = ws + o; o += (size_t)NCN * HID;
  float* vrel_c = ws + o; o += (size_t)NCN * HID;
  float* krel_r = ws + o; o += (size_t)NRN * HID;
  float* vrel_r = ws + o; o += (size_t)NRN * HID;
  float* agg_r  = ws + o; o += (size_t)NRN * HID;
  float* agg_c  = ws + o; o += (size_t)NCN * HID;
  float* alpha  = ws + o; o += (size_t)NEDGE * NHEAD;
  float* mbuf   = ws + o; o += (size_t)NCN * NHEAD;
  float* sbuf   = ws + o; o += (size_t)NCN * NHEAD;
  // scorer buffers alias the (dead after conv layers) kqv_c region
  float* zbuf   = kqv_c;                       // [NLBL,256]
  float* hmid   = kqv_c + (size_t)NLBL * 256;  // [NLBL,128]

  // ---- d_out layout: score | h_r | h_c ----
  float* out_score = (float*)d_out;
  float* out_hr    = out_score + NLBL;
  float* out_hc    = out_hr + (size_t)NRN * HID;

  // ---- input projection + relu ----
  gemm(stream, P.x_r, FIN, P.W_in_r, HID, P.b_in_r, h_r, HID,
       nullptr, 0, nullptr, NRN, HID, FIN, 0, 1);
  gemm(stream, P.x_c, FIN, P.W_in_c, HID, P.b_in_c, h_c, HID,
       nullptr, 0, nullptr, NCN, HID, FIN, 0, 1);

  const int eth = (NEDGE * NHEAD + 255) / 256;  // edge threads, per-(e,h)

  for (int l = 0; l < NLAYER; ++l) {
    const ConvP* cv = &P.conv[l];
    // KQV projections
    gemm(stream, h_r, HID, cv->Wkqv_r, 384, cv->bkqv_r, kqv_r, 384,
         nullptr, 0, nullptr, NRN, 384, HID, 0, 0);
    gemm(stream, h_c, HID, cv->Wkqv_c, 384, cv->bkqv_c, kqv_c, 384,
         nullptr, 0, nullptr, NCN, 384, HID, 0, 0);
    // per-head relation transforms: [N,32] @ [32,32] (strided views)
    for (int hd = 0; hd < NHEAD; ++hd) {
      const int ko = hd * DHEAD, vo = 256 + hd * DHEAD, wo = hd * DHEAD * DHEAD;
      gemm(stream, kqv_c + ko, 384, cv->a_c2r + wo, DHEAD, nullptr,
           krel_c + ko, HID, nullptr, 0, nullptr, NCN, DHEAD, DHEAD, 0, 0);
      gemm(stream, kqv_c + vo, 384, cv->m_c2r + wo, DHEAD, nullptr,
           vrel_c + ko, HID, nullptr, 0, nullptr, NCN, DHEAD, DHEAD, 0, 0);
      gemm(stream, kqv_r + ko, 384, cv->a_r2c + wo, DHEAD, nullptr,
           krel_r + ko, HID, nullptr, 0, nullptr, NRN, DHEAD, DHEAD, 0, 0);
      gemm(stream, kqv_r + vo, 384, cv->m_r2c + wo, DHEAD, nullptr,
           vrel_r + ko, HID, nullptr, 0, nullptr, NRN, DHEAD, DHEAD, 0, 0);
    }
    // ---- c2r attention (src = c nodes, dst = r nodes) ----
    {
      const int* src = P.edge_c2r;
      const int* dst = P.edge_c2r + NEDGE;
      fill(stream, mbuf, -INFINITY, (long)NRN * NHEAD);
      fill(stream, sbuf, 0.0f, (long)NRN * NHEAD);
      fill(stream, agg_r, 0.0f, (long)NRN * HID);
      edge_logits_kernel<<<eth, 256, 0, stream>>>(src, dst, krel_c, kqv_r,
                                                  cv->p_c2r, alpha, mbuf, NEDGE);
      edge_exp_kernel<<<eth, 256, 0, stream>>>(dst, alpha, mbuf, sbuf, NEDGE);
      edge_scatter_kernel<<<eth, 256, 0, stream>>>(src, dst, alpha, sbuf,
                                                   vrel_c, agg_r, NEDGE);
    }
    // ---- r2c attention (src = r nodes, dst = c nodes) ----
    {
      const int* src = P.edge_r2c;
      const int* dst = P.edge_r2c + NEDGE;
      fill(stream, mbuf, -INFINITY, (long)NCN * NHEAD);
      fill(stream, sbuf, 0.0f, (long)NCN * NHEAD);
      fill(stream, agg_c, 0.0f, (long)NCN * HID);
      edge_logits_kernel<<<eth, 256, 0, stream>>>(src, dst, krel_r, kqv_c,
                                                  cv->p_r2c, alpha, mbuf, NEDGE);
      edge_exp_kernel<<<eth, 256, 0, stream>>>(dst, alpha, mbuf, sbuf, NEDGE);
      edge_scatter_kernel<<<eth, 256, 0, stream>>>(src, dst, alpha, sbuf,
                                                   vrel_r, agg_c, NEDGE);
    }
    // ---- out_skip: h = sg * (gelu(agg) @ W_out + b) + (1-sg) * h ----
    float* tr = (l == NLAYER - 1) ? out_hr : h_r;
    float* tc = (l == NLAYER - 1) ? out_hc : h_c;
    gemm(stream, agg_r, HID, cv->Wout_r, HID, cv->bout_r, tr, HID,
         h_r, HID, cv->skip_r, NRN, HID, HID, 1, 2);
    gemm(stream, agg_c, HID, cv->Wout_c, HID, cv->bout_c, tc, HID,
         h_c, HID, cv->skip_c, NCN, HID, HID, 1, 2);
  }

  // ---- edge scorer ----
  {
    const int* row = P.label_edge;
    const int* col = P.label_edge + NLBL;
    long zn = (long)NLBL * 256;
    zgather_kernel<<<(int)((zn + 255) / 256), 256, 0, stream>>>(
        row, col, out_hr, out_hc, zbuf, NLBL);
    gemm(stream, zbuf, 256, P.W_mlp1, HID, P.b_mlp1, hmid, HID,
         nullptr, 0, nullptr, NLBL, HID, 256, 0, 1);
    score_kernel<<<(NLBL + 255) / 256, 256, 0, stream>>>(
        hmid, P.W_mlp2, P.b_mlp2, out_score, NLBL);
  }
}
